// RNN_38577396252937
// MI455X (gfx1250) — matrix operand order repaired and checked by reading, not compiled, not run on hardware
//
#include <hip/hip_runtime.h>

// ---------------------------------------------------------------------------
// LSTM forward for MI455X (gfx1250, wave32, WMMA).
//   tokens -> embed(bf16) -> fused [input GEMM + recurrent GEMM + cell] x 512
//   persistent kernel, c-state in registers, weight tiles staged in LDS
//   (48 KB/WG) to keep VGPR pressure < 256 (no spills, no vgpr-msb churn),
//   h double-buffered in L2, bf16 v_wmma_f32_16x16x32_bf16 with f32 accum.
// ---------------------------------------------------------------------------

typedef __bf16 bf16_t;
typedef __attribute__((ext_vector_type(16))) __bf16 v16bf;
typedef __attribute__((ext_vector_type(8)))  __bf16 v8bf;
typedef __attribute__((ext_vector_type(8)))  float  v8f;

#define VOCAB 50000
#define EMB   128
#define HID   256
#define G4    1024   // 4*HID
#define OUTD  64
#define BATCH 256
#define TLEN  512
#define NWG   16     // one WG per 16 h-units

#define NTILE_IH 16  // (EMB/32) * 4 gates
#define NTILE_HH 32  // (HID/32) * 4 gates
#define NTILE    (NTILE_IH + NTILE_HH)

#if __has_builtin(__builtin_amdgcn_tanhf)
__device__ __forceinline__ float fast_tanh(float x) { return __builtin_amdgcn_tanhf(x); }
#elif __has_builtin(__builtin_amdgcn_tanh_f32)
__device__ __forceinline__ float fast_tanh(float x) { return __builtin_amdgcn_tanh_f32(x); }
#else
__device__ __forceinline__ float fast_tanh(float x) { return tanhf(x); }
#endif

__device__ __forceinline__ float sigmoidf_(float x) {
    return 1.0f / (1.0f + __expf(-x));
}

__device__ __forceinline__ v8f wmma_bf16(v16bf a, v16bf b, v8f c) {
    // (neg_a, A, neg_b, B, c_mod, C, reuse_a, reuse_b)
    return __builtin_amdgcn_wmma_f32_16x16x32_bf16(false, a, false, b,
                                                   (short)0, c, false, false);
}

// A-fragment (16x32 bf16, M=rows striped as per ISA 7.12.2) from a row-major
// bf16 row pointer. lane<16: K {k0..k0+7, k0+16..k0+23}; lane>=16 shifted by 8.
__device__ __forceinline__ v16bf a_frag(const bf16_t* __restrict__ rowp,
                                        int k0, int half) {
    union { v16bf v; v8bf h[2]; } u;
    const bf16_t* r = rowp + k0 + half * 8;
    u.h[0] = *(const v8bf*)(r);
    u.h[1] = *(const v8bf*)(r + 16);
    return u.v;
}

// ---------------------------------------------------------------------------
// prep: last-valid index per row, bias_sum, zero h(-1) buffer, reset barrier
// ---------------------------------------------------------------------------
__global__ void prep_kernel(const int* __restrict__ x,
                            const float* __restrict__ b_ih,
                            const float* __restrict__ b_hh,
                            int* __restrict__ idx,
                            float* __restrict__ bias,
                            bf16_t* __restrict__ hbuf,
                            int* __restrict__ bar) {
    int tid = blockIdx.x * blockDim.x + threadIdx.x;
    if (tid == 0) *bar = 0;
    if (tid < BATCH) {
        int p = 0;
        for (int t = 0; t < TLEN; ++t)
            if (x[tid * TLEN + t] == 0) { p = t; break; }
        idx[tid] = p - 1;
    }
    if (tid < G4) bias[tid] = b_ih[tid] + b_hh[tid];
    // zero the "previous h" buffer (slot 1, read at t=0)
    for (int i = tid; i < BATCH * HID; i += gridDim.x * blockDim.x)
        hbuf[BATCH * HID + i] = (bf16_t)0.0f;
}

// f32 -> bf16 convert (embedding table)
__global__ void cvt_kernel(const float* __restrict__ src,
                           bf16_t* __restrict__ dst, int n) {
    for (int i = blockIdx.x * blockDim.x + threadIdx.x; i < n;
         i += gridDim.x * blockDim.x)
        dst[i] = (bf16_t)src[i];
}

// Pack W ([Ntot,Ktot] row-major f32, used as B = W^T [Ktot x Ntot]) into WMMA
// B-fragment order: tile(kt,nt) -> 512 bf16, lane-major, so a wave loads its
// fragment with one contiguous v16bf per lane.
// lane = half*16+n ; element e <-> B[kt*32 + (e<8 ? half*8+e : 16+half*8+e-8)][nt*16 + n]
__global__ void pack_w_kernel(const float* __restrict__ W,
                              bf16_t* __restrict__ dst, int Ktot, int Ntot) {
    int gt = blockIdx.x * blockDim.x + threadIdx.x;
    int lane = gt & 31;
    int tile = gt >> 5;
    int ntiles = Ntot / 16;
    if (tile >= (Ktot / 32) * ntiles) return;
    int kt = tile / ntiles, nt = tile % ntiles;
    int half = lane >> 4, n = lane & 15;
    int ncol = nt * 16 + n;
#pragma unroll
    for (int e = 0; e < 16; ++e) {
        int k = kt * 32 + (e < 8 ? half * 8 + e : 16 + half * 8 + (e - 8));
        dst[(size_t)tile * 512 + lane * 16 + e] = (bf16_t)W[ncol * Ktot + k];
    }
}

// ---------------------------------------------------------------------------
// Persistent LSTM kernel. Grid = 16 WGs x 256 threads (8 waves).
// WG cb owns h-units [cb*16, cb*16+16): gate cols {g*256 + cb*16} g=0..3.
// Wave w owns batch rows [w*32, w*32+32) (two 16-row WMMA tiles).
// Weight tiles for this WG live in LDS; c-state lives in registers.
// ---------------------------------------------------------------------------
__global__ __launch_bounds__(256)
void lstm_kernel(const int* __restrict__ x,
                 const bf16_t* __restrict__ emb_bf,
                 const bf16_t* __restrict__ wih_pk,   // [4][64] tiles
                 const bf16_t* __restrict__ whh_pk,   // [8][64] tiles
                 const float* __restrict__ bias,      // b_ih + b_hh
                 const int* __restrict__ idx,
                 bf16_t* __restrict__ hbuf,           // [2][BATCH][HID]
                 float* __restrict__ last,            // [BATCH][HID]
                 int* __restrict__ bar) {
    const int cb   = blockIdx.x;
    const int wave = threadIdx.x >> 5;
    const int lane = threadIdx.x & 31;
    const int half = lane >> 4;
    const int m    = lane & 15;
    const int rowBase = wave * 32;
    const int j = cb * 16 + m;           // owned h-unit (C/D col = lane%16)

    // ---- stage this WG's weight slice into LDS (once) ----
    // LDS tile layout: [kt*4+g] for W_ih, [16 + kt*4+g] for W_hh; 1KB each.
    __shared__ bf16_t lds_w[NTILE * 512];
    for (int i = threadIdx.x; i < NTILE_IH * 64; i += blockDim.x) {
        int tile = i >> 6;               // 64 x v8bf chunks per tile
        int off  = (i & 63) * 8;
        int kt = tile >> 2, g = tile & 3;
        *(v8bf*)&lds_w[tile * 512 + off] =
            *(const v8bf*)&wih_pk[((size_t)(kt * 64 + g * 16 + cb)) * 512 + off];
    }
    for (int i = threadIdx.x; i < NTILE_HH * 64; i += blockDim.x) {
        int tile = i >> 6;
        int off  = (i & 63) * 8;
        int kt = tile >> 2, g = tile & 3;
        *(v8bf*)&lds_w[(NTILE_IH + tile) * 512 + off] =
            *(const v8bf*)&whh_pk[((size_t)(kt * 64 + g * 16 + cb)) * 512 + off];
    }
    __syncthreads();

    float c_reg[2][8];
    int   myidx[2][8];
#pragma unroll
    for (int rt = 0; rt < 2; ++rt)
#pragma unroll
        for (int r = 0; r < 8; ++r) {
            c_reg[rt][r] = 0.0f;
            myidx[rt][r] = idx[rowBase + rt * 16 + half * 8 + r];
        }

    float bini[4];
#pragma unroll
    for (int g = 0; g < 4; ++g) bini[g] = bias[g * HID + j];

    for (int t = 0; t < TLEN; ++t) {
        const bf16_t* hprev = hbuf + (size_t)((t & 1) ^ 1) * (BATCH * HID);
        bf16_t*       hcur  = hbuf + (size_t)(t & 1) * (BATCH * HID);

        v8f acc[2][4];
#pragma unroll
        for (int rt = 0; rt < 2; ++rt)
#pragma unroll
            for (int g = 0; g < 4; ++g)
#pragma unroll
                for (int r = 0; r < 8; ++r) acc[rt][g][r] = bini[g];

        // gathered embedding row pointers for this timestep
        const bf16_t* arow0;
        const bf16_t* arow1;
        {
            int row0 = rowBase + m;
            int row1 = rowBase + 16 + m;
            arow0 = emb_bf + (size_t)x[row0 * TLEN + t] * EMB;
            arow1 = emb_bf + (size_t)x[row1 * TLEN + t] * EMB;
        }

        // ---- input GEMM contribution: K = EMB = 128 (4 k-tiles) ----
#pragma unroll
        for (int kt = 0; kt < EMB / 32; ++kt) {
            v16bf a0 = a_frag(arow0, kt * 32, half);
            v16bf a1 = a_frag(arow1, kt * 32, half);
#pragma unroll
            for (int g = 0; g < 4; ++g) {
                v16bf b = *(const v16bf*)&lds_w[(kt * 4 + g) * 512 + lane * 16];
                acc[0][g] = wmma_bf16(a0, b, acc[0][g]);
                acc[1][g] = wmma_bf16(a1, b, acc[1][g]);
            }
        }

        // ---- recurrent GEMM: K = HID = 256 (8 k-tiles) ----
#pragma unroll
        for (int kt = 0; kt < HID / 32; ++kt) {
            v16bf a0 = a_frag(hprev + (size_t)(rowBase + m) * HID, kt * 32, half);
            v16bf a1 = a_frag(hprev + (size_t)(rowBase + 16 + m) * HID, kt * 32, half);
#pragma unroll
            for (int g = 0; g < 4; ++g) {
                v16bf b = *(const v16bf*)&lds_w[(NTILE_IH + kt * 4 + g) * 512 + lane * 16];
                acc[0][g] = wmma_bf16(a0, b, acc[0][g]);
                acc[1][g] = wmma_bf16(a1, b, acc[1][g]);
            }
        }

        // ---- LSTM cell (all gates local), h -> bf16 global ----
#pragma unroll
        for (int rt = 0; rt < 2; ++rt)
#pragma unroll
            for (int r = 0; r < 8; ++r) {
                int row = rowBase + rt * 16 + half * 8 + r;
                float gi = acc[rt][0][r];
                float gf = acc[rt][1][r];
                float gg = acc[rt][2][r];
                float go = acc[rt][3][r];
                float c = sigmoidf_(gf) * c_reg[rt][r] +
                          sigmoidf_(gi) * fast_tanh(gg);
                c_reg[rt][r] = c;
                float h = sigmoidf_(go) * fast_tanh(c);
                hcur[(size_t)row * HID + j] = (bf16_t)h;
                if (myidx[rt][r] == t) last[(size_t)row * HID + j] = h;
            }

        // prefetch next timestep's embedding rows into cache hierarchy
        if (t + 1 < TLEN) {
            int row = rowBase + m;
            int tok = x[row * TLEN + (t + 1)];
            __builtin_prefetch(emb_bf + (size_t)tok * EMB, 0, 1);
        }

        // ---- grid-wide barrier: all h(t) visible before any h(t) read ----
        __syncthreads();
        if (threadIdx.x == 0) {
            __threadfence();
            __hip_atomic_fetch_add(bar, 1, __ATOMIC_RELEASE,
                                   __HIP_MEMORY_SCOPE_AGENT);
            const int target = NWG * (t + 1);
            while (__hip_atomic_load(bar, __ATOMIC_ACQUIRE,
                                     __HIP_MEMORY_SCOPE_AGENT) < target)
                __builtin_amdgcn_s_sleep(2);
        }
        __syncthreads();
    }
}

// logits = last @ W_out^T + b_out ; softmax over 64 (tiny, f32 scalar)
__global__ __launch_bounds__(64)
void out_kernel(const float* __restrict__ last,
                const float* __restrict__ W_out,
                const float* __restrict__ b_out,
                float* __restrict__ out) {
    int b = blockIdx.x, o = threadIdx.x;
    const float* h = last + (size_t)b * HID;
    const float* w = W_out + (size_t)o * HID;
    float acc = b_out[o];
    for (int k = 0; k < HID; ++k) acc += h[k] * w[k];
    __shared__ float sm[OUTD];
    sm[o] = acc;
    __syncthreads();
    float mx = sm[0];
    for (int k = 1; k < OUTD; ++k) mx = fmaxf(mx, sm[k]);
    __syncthreads();
    float e = __expf(acc - mx);
    sm[o] = e;
    __syncthreads();
    float s = 0.0f;
    for (int k = 0; k < OUTD; ++k) s += sm[k];
    out[(size_t)b * OUTD + o] = e / s;
}

// ---------------------------------------------------------------------------
extern "C" void kernel_launch(void* const* d_in, const int* in_sizes, int n_in,
                              void* d_out, int out_size, void* d_ws,
                              size_t ws_size, hipStream_t stream) {
    const int*   x     = (const int*)d_in[0];
    const float* emb   = (const float*)d_in[1];
    const float* W_ih  = (const float*)d_in[2];
    const float* W_hh  = (const float*)d_in[3];
    const float* b_ih  = (const float*)d_in[4];
    const float* b_hh  = (const float*)d_in[5];
    const float* W_out = (const float*)d_in[6];
    const float* b_out = (const float*)d_in[7];
    float* out = (float*)d_out;

    // workspace carve-up (~14.2 MB total)
    char* ws = (char*)d_ws;
    size_t off = 0;
    auto carve = [&](size_t bytes) -> void* {
        void* p = ws + off;
        off = (off + bytes + 255) & ~(size_t)255;
        return p;
    };
    int*    bar    = (int*)carve(16);
    int*    idx    = (int*)carve(BATCH * sizeof(int));
    float*  bias   = (float*)carve(G4 * sizeof(float));
    bf16_t* emb_bf = (bf16_t*)carve((size_t)VOCAB * EMB * sizeof(bf16_t));
    bf16_t* wih_pk = (bf16_t*)carve((size_t)(EMB / 32) * (G4 / 16) * 512 * sizeof(bf16_t));
    bf16_t* whh_pk = (bf16_t*)carve((size_t)(HID / 32) * (G4 / 16) * 512 * sizeof(bf16_t));
    bf16_t* hbuf   = (bf16_t*)carve((size_t)2 * BATCH * HID * sizeof(bf16_t));
    float*  last   = (float*)carve((size_t)BATCH * HID * sizeof(float));

    prep_kernel<<<256, 256, 0, stream>>>(x, b_ih, b_hh, idx, bias, hbuf, bar);
    cvt_kernel<<<512, 256, 0, stream>>>(emb, emb_bf, VOCAB * EMB);
    pack_w_kernel<<<(4 * 64 * 32) / 256, 256, 0, stream>>>(W_ih, wih_pk, EMB, G4);
    pack_w_kernel<<<(8 * 64 * 32) / 256, 256, 0, stream>>>(W_hh, whh_pk, HID, G4);
    lstm_kernel<<<NWG, 256, 0, stream>>>(x, emb_bf, wih_pk, whh_pk, bias, idx,
                                         hbuf, last, bar);
    out_kernel<<<BATCH, OUTD, 0, stream>>>(last, W_out, b_out, out);
}
